// HybridASAM_12120397709558
// MI455X (gfx1250) — matrix operand order repaired and checked by reading, not compiled, not run on hardware
//
#include <hip/hip_runtime.h>
#include <hip/hip_bf16.h>

// ---------------------------------------------------------------------------
// Hybrid local+strided-global attention for MI455X (gfx1250, wave32, WMMA).
// All heavy math on v_wmma_f32_16x16x32_f16 (f16 inputs, fp32 accumulate).
// GEMM staging now uses CDNA5 async memory->LDS (global_load_async_to_lds,
// ASYNCcnt) with double buffering: one barrier per K step, no VGPR round-trip.
// ---------------------------------------------------------------------------

typedef _Float16 h8_t  __attribute__((ext_vector_type(8)));
typedef _Float16 h16_t __attribute__((ext_vector_type(16)));
typedef float    f8_t  __attribute__((ext_vector_type(8)));
typedef __attribute__((address_space(3))) _Float16 lds_h;

#define SEQ     4096
#define NBATCH  2
#define NHEADS  16
#define DHEAD   64
#define DIM     1024
#define TOKENS  8192   // NBATCH * SEQ
#define QKVN    3072
#define LDK     48     // padded LDS row (halfs): 96B rows, 16B/32B aligned frags

__device__ __forceinline__ f8_t wmma16(h16_t a, h16_t b, f8_t c) {
  return __builtin_amdgcn_wmma_f32_16x16x32_f16(false, a, false, b, (short)0, c,
                                                false, false);
}

__device__ __forceinline__ h16_t cat8(h8_t a, h8_t b) {
  return __builtin_shufflevector(a, b, 0, 1, 2, 3, 4, 5, 6, 7, 8, 9, 10, 11, 12,
                                 13, 14, 15);
}

// A-operand fragment (16x32 f16): lane lo holds row; lanes 0-15
// carry K {0..15}, lanes 16-31 carry K {16..31}.
__device__ __forceinline__ h16_t loadA(const _Float16* rowPtr, int hi) {
  h8_t a0 = *(const h8_t*)(rowPtr + hi * 16);
  h8_t a1 = *(const h8_t*)(rowPtr + hi * 16 + 8);
  return cat8(a0, a1);
}

// ---- CDNA5 async memory->LDS (ISA 10.x / CDNA5 ISA), ASYNCcnt ----
__device__ __forceinline__ void async_copy_b128(const _Float16* g, _Float16* l) {
  unsigned lds_off = (unsigned)(size_t)(lds_h*)l;       // 32-bit LDS address
  unsigned long long ga = (unsigned long long)(size_t)g;  // 64-bit VGPR pair
  asm volatile("global_load_async_to_lds_b128 %0, %1, off"
               :: "v"(lds_off), "v"(ga)
               : "memory");
}

__device__ __forceinline__ void wait_async0() {
#if __has_builtin(__builtin_amdgcn_s_wait_asynccnt)
  __builtin_amdgcn_s_wait_asynccnt(0);
#else
  asm volatile("s_wait_asynccnt 0" ::: "memory");
#endif
}

// ------------------------------- prep kernels ------------------------------

__global__ void cvt_f32_f16(const float* __restrict__ in,
                            _Float16* __restrict__ out, int n) {
  int i = blockIdx.x * blockDim.x + threadIdx.x;
  if (i < n) out[i] = (_Float16)in[i];
}

// in: [K][N] fp32 row-major  ->  out: [N][K] f16 row-major (B^T for WMMA)
__global__ void transpose_to_f16(const float* __restrict__ in,
                                 _Float16* __restrict__ out, int K, int N) {
  int i = blockIdx.x * blockDim.x + threadIdx.x;
  if (i >= K * N) return;
  int k = i / N, n = i - k * N;
  out[(size_t)n * K + k] = (_Float16)in[i];
}

// gate[token] = sigmoid(x[token] . W_gate + b_gate); one wave per token
__global__ __launch_bounds__(256) void gate_kernel(
    const float* __restrict__ x, const float* __restrict__ wg,
    const float* __restrict__ bg, float* __restrict__ gate) {
  int tok  = blockIdx.x * 8 + (threadIdx.x >> 5);
  int lane = threadIdx.x & 31;
  const float* xr = x + (size_t)tok * DIM;
  float acc = 0.f;
  #pragma unroll
  for (int j = 0; j < DIM / 32; ++j) acc += xr[lane + 32 * j] * wg[lane + 32 * j];
  #pragma unroll
  for (int off = 16; off >= 1; off >>= 1) acc += __shfl_xor(acc, off, 32);
  if (lane == 0) gate[tok] = 1.0f / (1.0f + __expf(-(acc + bg[0])));
}

// out16 = f16( g*local + (1-g)*global )
__global__ void combine_kernel(const float* __restrict__ loc,
                               const float* __restrict__ glb,
                               const float* __restrict__ gate,
                               _Float16* __restrict__ out, int total) {
  int i = blockIdx.x * blockDim.x + threadIdx.x;
  if (i >= total) return;
  float g = gate[i >> 10];  // row = i / DIM
  out[i] = (_Float16)(g * loc[i] + (1.0f - g) * glb[i]);
}

// ------------------------------- GEMM (f16 WMMA) ---------------------------
// C[M,N] = A[M,K] * B[K,N], B given transposed (Bt[N,K], K contiguous).
// 256 threads = 8 waves, 128x128 block tile, wave tile 32x64 (2x4 WMMA tiles).
// K chunks of 32 are double-buffered in LDS via async loads (ASYNCcnt).
// MODE 0: QKV epilogue -> scatter f16 into q16/k16/v16t (+ strided k/v).
// MODE 1: out epilogue -> fp32 + bias into d_out.

template <int MODE>
__global__ __launch_bounds__(256) void gemm_f16(
    const _Float16* __restrict__ A, const _Float16* __restrict__ Bt, int K,
    _Float16* __restrict__ q16, _Float16* __restrict__ k16,
    _Float16* __restrict__ v16t, _Float16* __restrict__ kstr,
    _Float16* __restrict__ vstrT, float* __restrict__ outp,
    const float* __restrict__ bias) {
  __shared__ __align__(16) _Float16 lA[2][128 * LDK];
  __shared__ __align__(16) _Float16 lB[2][128 * LDK];
  const int tid  = threadIdx.x;
  const int lane = tid & 31;
  const int wave = tid >> 5;
  const int lo = lane & 15, hi = lane >> 4;
  const int wm = wave & 3, wn = wave >> 2;  // 4 (M) x 2 (N) wave grid
  const int blockM = blockIdx.y * 128;
  const int blockN = blockIdx.x * 128;

  f8_t acc[2][4] = {};

  // cooperative staging: 2 threads per row, 16 halfs (2 x b128) each
  const int ldr = tid >> 1;
  const int lco = (tid & 1) * 16;
  const _Float16* gA = A + (size_t)(blockM + ldr) * K + lco;
  const _Float16* gB = Bt + (size_t)(blockN + ldr) * K + lco;
  const int sOff = ldr * LDK + lco;

  auto stage = [&](int buf, int kc) {
    async_copy_b128(gA + kc,     &lA[buf][sOff]);
    async_copy_b128(gA + kc + 8, &lA[buf][sOff + 8]);
    async_copy_b128(gB + kc,     &lB[buf][sOff]);
    async_copy_b128(gB + kc + 8, &lB[buf][sOff + 8]);
  };

  stage(0, 0);
  wait_async0();
  __syncthreads();

  int cur = 0;
  for (int kc = 0; kc < K; kc += 32) {
    if (kc + 32 < K) stage(cur ^ 1, kc + 32);  // overlap next chunk with WMMA

    const _Float16* aRow = &lA[cur][(wm * 32 + lo) * LDK];
    const _Float16* bRow = &lB[cur][(wn * 64 + lo) * LDK + hi * 16];
    h16_t af[2];
    #pragma unroll
    for (int mt = 0; mt < 2; ++mt) af[mt] = loadA(aRow + mt * 16 * LDK, hi);
    #pragma unroll
    for (int nt = 0; nt < 4; ++nt) {
      h16_t bf = *(const h16_t*)(bRow + nt * 16 * LDK);
      #pragma unroll
      for (int mt = 0; mt < 2; ++mt) acc[mt][nt] = wmma16(af[mt], bf, acc[mt][nt]);
    }

    wait_async0();    // my async writes to buf cur^1 have landed in LDS
    __syncthreads();  // everyone done reading cur & done staging cur^1
    cur ^= 1;
  }

  #pragma unroll
  for (int mt = 0; mt < 2; ++mt)
    #pragma unroll
    for (int nt = 0; nt < 4; ++nt)
      #pragma unroll
      for (int i = 0; i < 8; ++i) {
        int r = blockM + wm * 32 + mt * 16 + hi * 8 + i;  // C layout: M=i+8*hi
        int c = blockN + wn * 64 + nt * 16 + lo;          // N = lane lo
        float v = acc[mt][nt][i];
        if (MODE == 1) {
          outp[(size_t)r * DIM + c] = v + bias[c];
        } else {
          _Float16 hv = (_Float16)v;
          int bb = r >> 12, ss = r & (SEQ - 1);
          int sect = c >> 10, n = c & (DIM - 1);
          int hh = n >> 6, d = n & 63;
          int bh = bb * NHEADS + hh;
          if (sect == 0) {
            q16[((size_t)bh * SEQ + ss) * DHEAD + d] = hv;
          } else if (sect == 1) {
            k16[((size_t)bh * SEQ + ss) * DHEAD + d] = hv;
            if ((ss & 31) == 0) kstr[((size_t)bh * 128 + (ss >> 5)) * DHEAD + d] = hv;
          } else {
            v16t[((size_t)bh * DHEAD + d) * SEQ + ss] = hv;
            if ((ss & 31) == 0) vstrT[((size_t)bh * DHEAD + d) * 128 + (ss >> 5)] = hv;
          }
        }
      }
}

// --------------------------- flash attention kernel ------------------------
// grid.x = NBATCH*NHEADS*8 ; 512 threads = 16 waves, each wave owns 32 queries.
// Local mode : 16 chunks of 64 keys at window offset w*512-256, masked to seq.
// Global mode: 2 chunks of 64 strided keys, never masked.

__global__ __launch_bounds__(512) void attn_kernel(
    const _Float16* __restrict__ qAll, const _Float16* __restrict__ kAll,
    const _Float16* __restrict__ vtAll, float* __restrict__ outp, int kRows,
    int nChunks, int isLocal) {
  __shared__ __align__(16) _Float16 pLds[16 * 32 * 64];  // 64 KB: P tiles/wave
  const int tid = threadIdx.x;
  const int lane = tid & 31, wave = tid >> 5;
  const int lo = lane & 15, hi = lane >> 4;
  const int bh = blockIdx.x >> 3;
  const int qblk = blockIdx.x & 7;
  const int b = bh >> 4, h = bh & 15;
  const int qStart = qblk * 512 + wave * 32;

  const _Float16* q  = qAll + ((size_t)bh * SEQ + qStart) * DHEAD;
  const _Float16* kp = kAll + (size_t)bh * kRows * DHEAD;
  const _Float16* vt = vtAll + (size_t)bh * DHEAD * kRows;
  _Float16* myP = pLds + wave * (32 * 64);
  const int keyOff = isLocal ? (qblk * 512 - 256) : 0;

  h16_t qf[2][2];  // 2 query tiles x 2 d-chunks, reused across all key chunks
  #pragma unroll
  for (int mt = 0; mt < 2; ++mt)
    #pragma unroll
    for (int dc = 0; dc < 2; ++dc)
      qf[mt][dc] = loadA(q + (mt * 16 + lo) * DHEAD + dc * 32, hi);

  f8_t o[2][4] = {};
  float mrun[2][8], lrun[2][8];
  #pragma unroll
  for (int mt = 0; mt < 2; ++mt)
    #pragma unroll
    for (int i = 0; i < 8; ++i) { mrun[mt][i] = -1e30f; lrun[mt][i] = 0.f; }

  for (int c = 0; c < nChunks; ++c) {
    const int cb = c * 64;

    // ---- S = Q K^T over this 64-key chunk (K-dim = DHEAD, 2 WMMAs) ----
    f8_t s[2][4] = {};
    #pragma unroll
    for (int nt = 0; nt < 4; ++nt) {
      int krow = keyOff + cb + nt * 16 + lo;                 // B lane = key col
      krow = krow < 0 ? 0 : (krow > kRows - 1 ? kRows - 1 : krow);
      const _Float16* kr = kp + (size_t)krow * DHEAD;
      h16_t k0 = *(const h16_t*)(kr + hi * 16);
      h16_t k1 = *(const h16_t*)(kr + 32 + hi * 16);
      #pragma unroll
      for (int mt = 0; mt < 2; ++mt) {
        s[mt][nt] = wmma16(qf[mt][0], k0, s[mt][nt]);
        s[mt][nt] = wmma16(qf[mt][1], k1, s[mt][nt]);
      }
    }

    // ---- scale + window mask ----
    #pragma unroll
    for (int nt = 0; nt < 4; ++nt) {
      int pos = keyOff + cb + nt * 16 + lo;
      bool valid = (pos >= 0) && (pos < kRows);
      #pragma unroll
      for (int mt = 0; mt < 2; ++mt)
        #pragma unroll
        for (int i = 0; i < 8; ++i)
          s[mt][nt][i] = valid ? s[mt][nt][i] * 0.125f : -1.0e9f;
    }

    // ---- online softmax (row = i + 8*hi; half-wave shfl reductions) ----
    #pragma unroll
    for (int mt = 0; mt < 2; ++mt)
      #pragma unroll
      for (int i = 0; i < 8; ++i) {
        float mx = fmaxf(fmaxf(s[mt][0][i], s[mt][1][i]),
                         fmaxf(s[mt][2][i], s[mt][3][i]));
        mx = fmaxf(mx, __shfl_xor(mx, 8, 16));
        mx = fmaxf(mx, __shfl_xor(mx, 4, 16));
        mx = fmaxf(mx, __shfl_xor(mx, 2, 16));
        mx = fmaxf(mx, __shfl_xor(mx, 1, 16));
        float nm = fmaxf(mrun[mt][i], mx);
        float corr = __expf(mrun[mt][i] - nm);
        mrun[mt][i] = nm;
        float rs = 0.f;
        #pragma unroll
        for (int nt = 0; nt < 4; ++nt) {
          float p = __expf(s[mt][nt][i] - nm);
          s[mt][nt][i] = p;
          rs += p;
        }
        rs += __shfl_xor(rs, 8, 16);
        rs += __shfl_xor(rs, 4, 16);
        rs += __shfl_xor(rs, 2, 16);
        rs += __shfl_xor(rs, 1, 16);
        lrun[mt][i] = lrun[mt][i] * corr + rs;
        #pragma unroll
        for (int nt = 0; nt < 4; ++nt) o[mt][nt][i] *= corr;
      }

    // ---- P: C-fragment -> LDS (f16, XOR-swizzled 8-half groups) ----
    #pragma unroll
    for (int mt = 0; mt < 2; ++mt)
      #pragma unroll
      for (int nt = 0; nt < 4; ++nt)
        #pragma unroll
        for (int i = 0; i < 8; ++i) {
          int row = mt * 16 + hi * 8 + i;
          int col = nt * 16 + lo;
          int phys = (((col >> 3) ^ (row & 7)) << 3) + (col & 7);
          myP[row * 64 + phys] = (_Float16)s[mt][nt][i];
        }

    // ---- P A-fragments from LDS ----
    h16_t pf[2][2];
    #pragma unroll
    for (int mt = 0; mt < 2; ++mt) {
      int row = mt * 16 + lo;
      const _Float16* pr = myP + row * 64;
      int sw = row & 7;
      #pragma unroll
      for (int kc = 0; kc < 2; ++kc) {
        int t0 = kc * 32 + hi * 16;
        int t1 = kc * 32 + hi * 16 + 8;
        h8_t a0 = *(const h8_t*)(pr + (((t0 >> 3) ^ sw) << 3));
        h8_t a1 = *(const h8_t*)(pr + (((t1 >> 3) ^ sw) << 3));
        pf[mt][kc] = cat8(a0, a1);
      }
    }

    // ---- O += P * V  (V^T rows are d, keys contiguous) ----
    #pragma unroll
    for (int nt = 0; nt < 4; ++nt)
      #pragma unroll
      for (int kc = 0; kc < 2; ++kc) {
        int tb = keyOff + cb + kc * 32 + hi * 16;  // clamp run; masked P == 0
        tb = tb < 0 ? 0 : (tb > kRows - 16 ? kRows - 16 : tb);
        h16_t vf = *(const h16_t*)(vt + (size_t)(nt * 16 + lo) * kRows + tb);
        #pragma unroll
        for (int mt = 0; mt < 2; ++mt) o[mt][nt] = wmma16(pf[mt][kc], vf, o[mt][nt]);
      }
  }

  // ---- normalize + store fp32 [token, h*64+d] ----
  #pragma unroll
  for (int mt = 0; mt < 2; ++mt)
    #pragma unroll
    for (int i = 0; i < 8; ++i) {
      float inv = 1.0f / lrun[mt][i];
      int srow = qStart + mt * 16 + hi * 8 + i;
      #pragma unroll
      for (int nt = 0; nt < 4; ++nt) {
        int col = h * DHEAD + nt * 16 + lo;
        outp[((size_t)b * SEQ + srow) * DIM + col] = o[mt][nt][i] * inv;
      }
    }
}

// --------------------------------- launcher --------------------------------

extern "C" void kernel_launch(void* const* d_in, const int* in_sizes, int n_in,
                              void* d_out, int out_size, void* d_ws,
                              size_t ws_size, hipStream_t stream) {
  (void)in_sizes; (void)n_in; (void)out_size; (void)ws_size;
  const float* x    = (const float*)d_in[0];
  const float* Wqkv = (const float*)d_in[1];
  const float* Wg   = (const float*)d_in[2];
  const float* bg   = (const float*)d_in[3];
  const float* Wout = (const float*)d_in[4];
  const float* bout = (const float*)d_in[5];
  float* out = (float*)d_out;

  char* ws = (char*)d_ws;
  size_t off = 0;
  auto take = [&](size_t bytes) {
    char* p = ws + off;
    off += (bytes + 255) & ~(size_t)255;
    return p;
  };

  // Region A (33.5 MB): phase 1 holds x16 + WqkvT; dead after GEMM1, then
  // reused entirely as local32 (stream order guarantees no overlap in time).
  char* regionA = take((size_t)TOKENS * DIM * 4);
  _Float16* x16    = (_Float16*)regionA;
  _Float16* wqkvT  = (_Float16*)(regionA + (size_t)TOKENS * DIM * 2);
  float*    local32 = (float*)regionA;

  float*    glob32 = (float*)take((size_t)TOKENS * DIM * 4);
  _Float16* q16    = (_Float16*)take((size_t)TOKENS * DIM * 2);  // later comb16
  _Float16* k16    = (_Float16*)take((size_t)TOKENS * DIM * 2);
  _Float16* v16t   = (_Float16*)take((size_t)TOKENS * DIM * 2);
  _Float16* kstr   = (_Float16*)take((size_t)NBATCH * NHEADS * 128 * DHEAD * 2);
  _Float16* vstrT  = (_Float16*)take((size_t)NBATCH * NHEADS * 128 * DHEAD * 2);
  _Float16* woutT  = (_Float16*)take((size_t)DIM * DIM * 2);
  float*    gate   = (float*)take((size_t)TOKENS * 4);

  const int total = TOKENS * DIM;

  cvt_f32_f16<<<total / 256, 256, 0, stream>>>(x, x16, total);
  transpose_to_f16<<<(DIM * QKVN) / 256, 256, 0, stream>>>(Wqkv, wqkvT, DIM, QKVN);
  transpose_to_f16<<<(DIM * DIM) / 256, 256, 0, stream>>>(Wout, woutT, DIM, DIM);
  gate_kernel<<<TOKENS / 8, 256, 0, stream>>>(x, Wg, bg, gate);

  // QKV GEMM with layout-scattering epilogue
  gemm_f16<0><<<dim3(QKVN / 128, TOKENS / 128), 256, 0, stream>>>(
      x16, wqkvT, DIM, q16, k16, v16t, kstr, vstrT, nullptr, nullptr);

  // local windowed attention: 2*16*8 blocks, 1024 keys/window, masked
  attn_kernel<<<NBATCH * NHEADS * 8, 512, 0, stream>>>(q16, k16, v16t, local32,
                                                       SEQ, 16, 1);
  // sparse global attention: 128 strided keys, unmasked
  attn_kernel<<<NBATCH * NHEADS * 8, 512, 0, stream>>>(q16, kstr, vstrT, glob32,
                                                       128, 2, 0);

  // gate blend -> f16 (reuse q16 buffer as comb16)
  combine_kernel<<<total / 256, 256, 0, stream>>>(local32, glob32, gate, q16,
                                                  total);

  // output GEMM + bias
  gemm_f16<1><<<dim3(DIM / 128, TOKENS / 128), 256, 0, stream>>>(
      q16, woutT, DIM, nullptr, nullptr, nullptr, nullptr, nullptr, out, bout);
}
